// MixtureOfExperts_89653147337498
// MI455X (gfx1250) — matrix operand order repaired and checked by reading, not compiled, not run on hardware
//
#include <hip/hip_runtime.h>
#include <hip/hip_bf16.h>

#define TOK 2048
#define DIM 768
#define FFD 3072
#define NE 8
#define FC 128     // FF chunk width
#define KS 32      // WMMA K step (bf16)
#define NSLOT (2 * TOK)

typedef __attribute__((ext_vector_type(16))) __bf16 v16bf;
typedef __attribute__((ext_vector_type(8)))  __bf16 v8bf;
typedef __attribute__((ext_vector_type(2)))  __bf16 v2bf;
typedef __attribute__((ext_vector_type(8)))  float  v8f;
typedef __attribute__((ext_vector_type(4)))  float  v4f;
typedef __attribute__((ext_vector_type(4)))  int    v4i;

#define __AS1 __attribute__((address_space(1)))
#define __AS3 __attribute__((address_space(3)))

#if defined(__has_builtin)
#  if __has_builtin(__builtin_amdgcn_global_load_async_to_lds_b128)
#    define HAVE_ASYNC_LDS 1
#  endif
#endif
#ifndef HAVE_ASYNC_LDS
#  define HAVE_ASYNC_LDS 0
#endif

__device__ __forceinline__ v16bf cat16(v8bf lo, v8bf hi) {
  return __builtin_shufflevector(lo, hi, 0, 1, 2, 3, 4, 5, 6, 7,
                                 8, 9, 10, 11, 12, 13, 14, 15);
}

__device__ __forceinline__ float silu_f(float v) {
  return v / (1.0f + __expf(-v));
}

// ---------------------------------------------------------------- init
__global__ void moe_init_kernel(int* counts, float* eload) {
  int i = threadIdx.x;
  if (i < NE) { counts[i] = 0; eload[i] = 0.0f; }
}

// -------------------------------------------------------------- router
__global__ __launch_bounds__(256) void moe_router_kernel(
    const float* __restrict__ x, const float* __restrict__ rw,
    const float* __restrict__ rb, const float* __restrict__ dgw,
    const float* __restrict__ dgb, void* __restrict__ xb_,
    int* __restrict__ counts, int* __restrict__ perm,
    int* __restrict__ slot_e, float* __restrict__ slot_c,
    float* __restrict__ eload) {
  __bf16* xb = (__bf16*)xb_;
  int t = blockIdx.x * blockDim.x + threadIdx.x;
  if (t >= TOK) return;

  float acc[NE], dga[NE];
#pragma unroll
  for (int e = 0; e < NE; e++) { acc[e] = 0.0f; dga[e] = 0.0f; }

  const float* xr = x + (size_t)t * DIM;
  for (int d = 0; d < DIM; d++) {
    float xv = xr[d];
    xb[(size_t)t * DIM + d] = (__bf16)xv;
#pragma unroll
    for (int e = 0; e < NE; e++) {
      acc[e] += xv * rw[d * NE + e];
      dga[e] += xv * dgw[e * DIM + d];
    }
  }

  float mx = -1e30f;
#pragma unroll
  for (int e = 0; e < NE; e++) { acc[e] += rb[e]; mx = fmaxf(mx, acc[e]); }
  float p[NE], sum = 0.0f;
#pragma unroll
  for (int e = 0; e < NE; e++) { p[e] = __expf(acc[e] - mx); sum += p[e]; }
  float inv = 1.0f / sum;
#pragma unroll
  for (int e = 0; e < NE; e++) p[e] *= inv;

  int i1 = 0; float m1 = p[0];
#pragma unroll
  for (int e = 1; e < NE; e++) if (p[e] > m1) { m1 = p[e]; i1 = e; }
  int i2 = (i1 == 0) ? 1 : 0; float m2 = -1e30f;
#pragma unroll
  for (int e = 0; e < NE; e++)
    if (e != i1 && p[e] > m2) { m2 = p[e]; i2 = e; }

  float mm = fmaxf(m1, m2);
  float ea = __expf(m1 - mm), eb = __expf(m2 - mm);
  float tw1 = ea / (ea + eb), tw2 = eb / (ea + eb);
  float dg1 = 1.0f / (1.0f + __expf(-(dga[i1] + dgb[i1])));
  float dg2 = 1.0f / (1.0f + __expf(-(dga[i2] + dgb[i2])));

  slot_e[2 * t]     = i1;  slot_c[2 * t]     = dg1 * tw1;
  slot_e[2 * t + 1] = i2;  slot_c[2 * t + 1] = dg2 * tw2;

  int p1 = atomicAdd(&counts[i1], 1);
  perm[i1 * NSLOT + p1] = 2 * t;
  int p2 = atomicAdd(&counts[i2], 1);
  perm[i2 * NSLOT + p2] = 2 * t + 1;
  atomicAdd(&eload[i1], 1.0f);
  atomicAdd(&eload[i2], 1.0f);
}

// ------------------------------------------------------- expert FFN (WMMA)
// dynamic LDS layout (bf16 elements):
//   xs   [16*DIM]   24 KB   x tile
//   w1t  [FC*KS]     8 KB   w1 K-block transposed [col][k]
//   w3t  [FC*KS]     8 KB
//   acts [16*FC]     4 KB
//   w2t  [DIM*KS]   48 KB   w2 K-block transposed [col][k]
#define SMEM_ELEMS (16 * DIM + 2 * FC * KS + 16 * FC + DIM * KS)
#define SMEM_BYTES (SMEM_ELEMS * 2)

__global__ __launch_bounds__(256) void moe_ffn_kernel(
    const float* __restrict__ w1, const float* __restrict__ w1b,
    const float* __restrict__ w2, const float* __restrict__ w2b,
    const float* __restrict__ w3, const float* __restrict__ w3b,
    const void* __restrict__ xb_, const int* __restrict__ counts,
    const int* __restrict__ perm, float* __restrict__ ffout) {
  const __bf16* xb = (const __bf16*)xb_;
  const int e = blockIdx.y;
  const int tile = blockIdx.x;
  const int n = counts[e];
  if (tile * 16 >= n) return;

  extern __shared__ __align__(16) char smem[];
  __bf16* xs   = (__bf16*)smem;
  __bf16* w1t  = xs + 16 * DIM;
  __bf16* w3t  = w1t + FC * KS;
  __bf16* acts = w3t + FC * KS;
  __bf16* w2t  = acts + 16 * FC;
  __shared__ int slotS[16];
  __shared__ int valS[16];

  const int tid = threadIdx.x;
  if (tid < 16) {
    int idx = tile * 16 + tid;
    int ci = idx < n ? idx : n - 1;
    slotS[tid] = perm[e * NSLOT + ci];
    valS[tid] = (idx < n) ? 1 : 0;
  }
  __syncthreads();

  // ---- stage the 16 token rows (bf16) into LDS: 16*96 16-byte chunks ----
#if HAVE_ASYNC_LDS
#pragma unroll
  for (int it = 0; it < 6; ++it) {
    int i = tid + it * 256;
    int r = i / 96;
    int c = (i % 96) * 8;
    const __bf16* g = &xb[(size_t)(slotS[r] >> 1) * DIM + c];
    __AS1 v4i* gp = (__AS1 v4i*)(uintptr_t)g;
    __AS3 v4i* lp = (__AS3 v4i*)(unsigned)(uintptr_t)&xs[r * DIM + c];
    __builtin_amdgcn_global_load_async_to_lds_b128(gp, lp, 0, 0);
  }
#if __has_builtin(__builtin_amdgcn_s_wait_asynccnt)
  __builtin_amdgcn_s_wait_asynccnt(0);
#else
  asm volatile("s_wait_asynccnt 0" ::: "memory");
#endif
#else
#pragma unroll
  for (int it = 0; it < 6; ++it) {
    int i = tid + it * 256;
    int r = i / 96;
    int c = (i % 96) * 8;
    int tok = slotS[r] >> 1;
    *reinterpret_cast<uint4*>(&xs[r * DIM + c]) =
        *reinterpret_cast<const uint4*>(&xb[(size_t)tok * DIM + c]);
  }
#endif
  __syncthreads();

  const int wave = tid >> 5;
  const int lane = tid & 31;
  const int lrow = lane & 15;
  const int lhi = lane >> 4;  // 0/1: which K-half this lane holds

  const float* W1e = w1 + (size_t)e * DIM * FFD;
  const float* W3e = w3 + (size_t)e * DIM * FFD;
  const float* W2e = w2 + (size_t)e * FFD * DIM;

  v8f facc[6];  // 16 x 96 f32 output columns owned by this wave
#pragma unroll
  for (int i = 0; i < 6; i++)
#pragma unroll
    for (int j = 0; j < 8; j++) facc[i][j] = 0.0f;

  for (int f0 = 0; f0 < FFD; f0 += FC) {
    v8f hacc, gacc;
#pragma unroll
    for (int j = 0; j < 8; j++) { hacc[j] = 0.0f; gacc[j] = 0.0f; }

    // ---- GEMM1: h = x @ w1[:, f0:f0+FC], g = x @ w3[:, f0:f0+FC] ----
    for (int k0 = 0; k0 < DIM; k0 += KS) {
      // stage w1/w3 K-block transposed; pack (k,k+1) pairs into b32 stores.
      // 512 work items = 16 k-pairs x 32 col-quads; 2 per thread.
#pragma unroll
      for (int it = 0; it < 2; ++it) {
        int idx = tid + it * 256;
        int k = (idx >> 5) * 2;
        int col = (idx & 31) * 4;
        const float* g1 = &W1e[(size_t)(k0 + k) * FFD + f0 + col];
        const float* g3 = &W3e[(size_t)(k0 + k) * FFD + f0 + col];
        v4f a0 = *(const v4f*)g1;
        v4f a1 = *(const v4f*)(g1 + FFD);
        v4f b0 = *(const v4f*)g3;
        v4f b1 = *(const v4f*)(g3 + FFD);
#pragma unroll
        for (int j = 0; j < 4; j++) {
          v2bf ta; ta[0] = (__bf16)a0[j]; ta[1] = (__bf16)a1[j];
          v2bf tb; tb[0] = (__bf16)b0[j]; tb[1] = (__bf16)b1[j];
          *(v2bf*)&w1t[(col + j) * KS + k] = ta;
          *(v2bf*)&w3t[(col + j) * KS + k] = tb;
        }
      }
      if (k0 + KS < DIM)
        __builtin_prefetch(&W1e[(size_t)(k0 + KS) * FFD + f0 + (tid & 127)], 0, 1);
      __syncthreads();

      // A fragment (16x32 bf16): lane row = lane&15; K halves by lane>=16
      v16bf afr;
      {
        const __bf16* p = &xs[lrow * DIM + k0 + lhi * 16];
        afr = cat16(*reinterpret_cast<const v8bf*>(p),
                    *reinterpret_cast<const v8bf*>(p + 8));
      }
      // B fragments (32x16 bf16): lane col = lane&15, K half = (lane>>4)*16
      const int c = wave * 16 + lrow;
      v16bf b1f, b3f;
      {
        const __bf16* p = &w1t[c * KS + lhi * 16];
        b1f = cat16(*reinterpret_cast<const v8bf*>(p),
                    *reinterpret_cast<const v8bf*>(p + 8));
        const __bf16* q = &w3t[c * KS + lhi * 16];
        b3f = cat16(*reinterpret_cast<const v8bf*>(q),
                    *reinterpret_cast<const v8bf*>(q + 8));
      }
      hacc = __builtin_amdgcn_wmma_f32_16x16x32_bf16(false, afr, false, b1f,
                                                     (short)0, hacc, false, false);
      gacc = __builtin_amdgcn_wmma_f32_16x16x32_bf16(false, afr, false, b3f,
                                                     (short)0, gacc, false, false);
      __syncthreads();
    }

    // ---- epilogue: bias + silu(h1)*silu(g) -> acts (bf16) in LDS ----
    {
      int cg = f0 + wave * 16 + lrow;
      float b1v = w1b[e * FFD + cg];
      float b3v = w3b[e * FFD + cg];
#pragma unroll
      for (int r = 0; r < 8; r++) {
        float h = hacc[r] + b1v;
        float g = gacc[r] + b3v;
        float a = silu_f(h) * silu_f(g);
        int m = r + lhi * 8;  // C/D layout: VGPR r, lanes>=16 -> rows 8..15
        acts[m * FC + wave * 16 + lrow] = (__bf16)a;
      }
    }
    __syncthreads();

    // ---- GEMM2: ff += act(16xFC) @ w2[f0:f0+FC, :DIM] ----
#pragma unroll 1
    for (int k2 = 0; k2 < FC; k2 += KS) {
      // stage w2 K-block (32 x 768) transposed into w2t[col][k].
      // 3072 work items = 16 k-pairs x 192 col-quads; 12 per thread.
#pragma unroll
      for (int it = 0; it < 12; ++it) {
        int idx = tid + it * 256;
        int col = (idx % 192) * 4;
        int k = (idx / 192) * 2;
        const float* g0 = &W2e[(size_t)(f0 + k2 + k) * DIM + col];
        v4f a0 = *(const v4f*)g0;
        v4f a1 = *(const v4f*)(g0 + DIM);
#pragma unroll
        for (int j = 0; j < 4; j++) {
          v2bf t; t[0] = (__bf16)a0[j]; t[1] = (__bf16)a1[j];
          *(v2bf*)&w2t[(col + j) * KS + k] = t;
        }
      }
      if (k2 + KS < FC)
        __builtin_prefetch(&W2e[(size_t)(f0 + k2 + KS) * DIM + (tid & 255)], 0, 1);
      __syncthreads();

      v16bf afr2;
      {
        const __bf16* p = &acts[lrow * FC + k2 + lhi * 16];
        afr2 = cat16(*reinterpret_cast<const v8bf*>(p),
                     *reinterpret_cast<const v8bf*>(p + 8));
      }
#pragma unroll
      for (int nt = 0; nt < 6; nt++) {
        int col = wave * 96 + nt * 16 + lrow;
        const __bf16* p = &w2t[col * KS + lhi * 16];
        v16bf bfr = cat16(*reinterpret_cast<const v8bf*>(p),
                          *reinterpret_cast<const v8bf*>(p + 8));
        facc[nt] = __builtin_amdgcn_wmma_f32_16x16x32_bf16(
            false, afr2, false, bfr, (short)0, facc[nt], false, false);
      }
      __syncthreads();
    }
  }

  // ---- write ff output (+ w2 bias), masked for padding rows ----
#pragma unroll
  for (int nt = 0; nt < 6; nt++) {
    int col = wave * 96 + nt * 16 + lrow;
    float bv = w2b[e * DIM + col];
#pragma unroll
    for (int r = 0; r < 8; r++) {
      int m = r + lhi * 8;
      if (valS[m])
        ffout[(size_t)slotS[m] * DIM + col] = facc[nt][r] + bv;
    }
  }
}

// ----------------------------------------------- residual + LN + combine
__global__ __launch_bounds__(256) void moe_combine_kernel(
    const float* __restrict__ x, const int* __restrict__ slot_e,
    const float* __restrict__ slot_c, const float* __restrict__ ffout,
    const float* __restrict__ lng, const float* __restrict__ lnb,
    float* __restrict__ out) {
  int t = blockIdx.x;
  int tid = threadIdx.x;
  __shared__ float red[256];

  float o[3] = {0.0f, 0.0f, 0.0f};
  for (int k = 0; k < 2; k++) {
    int s = 2 * t + k;
    int e = slot_e[s];
    float coef = slot_c[s];
    float y[3];
    float lsum = 0.0f, lsq = 0.0f;
#pragma unroll
    for (int j = 0; j < 3; j++) {
      int d = tid + j * 256;
      float v = x[(size_t)t * DIM + d] + coef * ffout[(size_t)s * DIM + d];
      y[j] = v;
      lsum += v;
      lsq += v * v;
    }
    red[tid] = lsum;
    __syncthreads();
    for (int st = 128; st > 0; st >>= 1) {
      if (tid < st) red[tid] += red[tid + st];
      __syncthreads();
    }
    float mu = red[0] * (1.0f / DIM);
    __syncthreads();
    red[tid] = lsq;
    __syncthreads();
    for (int st = 128; st > 0; st >>= 1) {
      if (tid < st) red[tid] += red[tid + st];
      __syncthreads();
    }
    float var = red[0] * (1.0f / DIM) - mu * mu;
    __syncthreads();
    float rstd = rsqrtf(var + 1e-5f);
#pragma unroll
    for (int j = 0; j < 3; j++) {
      int d = tid + j * 256;
      o[j] += (y[j] - mu) * rstd * lng[(size_t)e * DIM + d] + lnb[(size_t)e * DIM + d];
    }
  }
#pragma unroll
  for (int j = 0; j < 3; j++) out[(size_t)t * DIM + tid + j * 256] = o[j];
}

// ---------------------------------------------------------------- launch
extern "C" void kernel_launch(void* const* d_in, const int* in_sizes, int n_in,
                              void* d_out, int out_size, void* d_ws, size_t ws_size,
                              hipStream_t stream) {
  const float* x   = (const float*)d_in[0];
  const float* rw  = (const float*)d_in[1];
  const float* rb  = (const float*)d_in[2];
  const float* dgw = (const float*)d_in[3];
  const float* dgb = (const float*)d_in[4];
  const float* w1  = (const float*)d_in[5];
  const float* w1b = (const float*)d_in[6];
  const float* w2  = (const float*)d_in[7];
  const float* w2b = (const float*)d_in[8];
  const float* w3  = (const float*)d_in[9];
  const float* w3b = (const float*)d_in[10];
  const float* lng = (const float*)d_in[11];
  const float* lnb = (const float*)d_in[12];

  float* out = (float*)d_out;
  char* ws = (char*)d_ws;

  // workspace layout (bytes)
  void*  xb     = (void*)(ws + 0);                 // TOK*DIM bf16 = 3,145,728
  int*   counts = (int*)(ws + 3145728);            // 32
  int*   perm   = (int*)(ws + 3145760);            // NE*NSLOT*4 = 131,072
  int*   slot_e = (int*)(ws + 3276832);            // NSLOT*4 = 16,384
  float* slot_c = (float*)(ws + 3293216);          // 16,384
  float* ffout  = (float*)(ws + 3309600);          // NSLOT*DIM*4 = 12,582,912
  float* eload  = out + (size_t)TOK * DIM;         // last NE floats of output

  hipLaunchKernelGGL(moe_init_kernel, dim3(1), dim3(64), 0, stream, counts, eload);
  hipLaunchKernelGGL(moe_router_kernel, dim3(TOK / 256), dim3(256), 0, stream,
                     x, rw, rb, dgw, dgb, xb, counts, perm, slot_e, slot_c, eload);
  hipLaunchKernelGGL(moe_ffn_kernel, dim3(NSLOT / 16, NE), dim3(256), SMEM_BYTES, stream,
                     w1, w1b, w2, w2b, w3, w3b, xb, counts, perm, ffout);
  hipLaunchKernelGGL(moe_combine_kernel, dim3(TOK), dim3(256), 0, stream,
                     x, slot_e, slot_c, ffout, lng, lnb, out);
}
